// MultiHeadAttentionLayer_30640296690508
// MI455X (gfx1250) — matrix operand order repaired and checked by reading, not compiled, not run on hardware
//
#include <hip/hip_runtime.h>
#include <hip/hip_bf16.h>

// Problem constants (from reference): N=2, L=2048, E=1024, H=16, DH=64
#define NB  2
#define LQ  2048
#define EMB 1024
#define NH  16
#define DH  64

typedef __bf16 bf16_t;
typedef __attribute__((ext_vector_type(16))) __bf16 v16bf;
typedef __attribute__((ext_vector_type(8)))  float  v8f;

__device__ __forceinline__ v8f wmma_bf16(v16bf a, v16bf b, v8f c) {
  // D = A(16x32) x B(32x16) + C, f32 accumulate
  return __builtin_amdgcn_wmma_f32_16x16x32_bf16(false, a, false, b,
                                                 (short)0, c, false, false);
}

// CDNA5 async global->LDS copy (ASYNCcnt tracked).  The LDS destination
// operand is the raw LDS byte offset; flat shared pointers truncate to the
// LDS offset in their low 32 bits (ISA 10.2 aperture rules).
__device__ __forceinline__ void async_copy_b128(void* lds_dst, const void* gsrc) {
  asm volatile("global_load_async_to_lds_b128 %0, %1, off"
               :: "v"((unsigned int)(unsigned long long)lds_dst),
                  "v"((unsigned long long)gsrc)
               : "memory");
}
__device__ __forceinline__ void wait_async0() {
  asm volatile("s_wait_asynccnt 0x0" ::: "memory");
}

// ---------------------------------------------------------------------------
// Kernel 1: per-head QKV projection.  Out = X(16x64) @ W^T(64x64) + b, bf16.
// grid = (N*L/128, H, 3[q,k,v]); block = 256 (8 waves); each wave: 16x64 tile.
// ---------------------------------------------------------------------------
__global__ __launch_bounds__(256)
void qkv_proj_kernel(const float* __restrict__ EPq, const float* __restrict__ EPk,
                     const float* __restrict__ EPv,
                     const float* __restrict__ Wq, const float* __restrict__ bq,
                     const float* __restrict__ Wk, const float* __restrict__ bk,
                     const float* __restrict__ Wv, const float* __restrict__ bv,
                     bf16_t* __restrict__ Qb, bf16_t* __restrict__ Kb,
                     bf16_t* __restrict__ Vb)
{
  const int typ = blockIdx.z;
  const float* EP  = (typ == 0) ? EPq : (typ == 1) ? EPk : EPv;
  const float* W   = (typ == 0) ? Wq  : (typ == 1) ? Wk  : Wv;
  const float* bia = (typ == 0) ? bq  : (typ == 1) ? bk  : bv;
  bf16_t*      Out = (typ == 0) ? Qb  : (typ == 1) ? Kb  : Vb;

  const int h       = blockIdx.y;
  const int rowBase = blockIdx.x * 128;          // flat row = n*L + l

  __shared__ __align__(16) bf16_t Ws[64][72];    // W, row major, bf16
  __shared__ __align__(16) bf16_t Xs[128][72];   // X tile, bf16

  for (int i = threadIdx.x; i < 64 * 64; i += 256) {
    int r = i >> 6, c = i & 63;
    Ws[r][c] = (bf16_t)W[i];
  }
  for (int i = threadIdx.x; i < 128 * 64; i += 256) {
    int r = i >> 6, c = i & 63;
    Xs[r][c] = (bf16_t)EP[(size_t)(rowBase + r) * EMB + h * DH + c];
  }
  __syncthreads();

  const int wave  = threadIdx.x >> 5;
  const int lane  = threadIdx.x & 31;
  const int rHalf = lane >> 4;                   // 0: lanes 0-15, 1: lanes 16-31
  const int lcol  = lane & 15;
  const int arow  = wave * 16 + lcol;

  // A fragments: lane holds row lane%16; K = h + 16*rHalf
  v16bf a0, a1;
  #pragma unroll
  for (int hh = 0; hh < 16; ++hh) {
    int k = hh + 16 * rHalf;
    a0[hh] = Xs[arow][k];
    a1[hh] = Xs[arow][32 + k];
  }

  #pragma unroll
  for (int nt = 0; nt < 4; ++nt) {
    // B = W^T tile: B[kk][nn] = W[nt*16+nn][kk]; lane holds col lane%16,
    // K = h + 16*rHalf
    v16bf b0, b1;
    #pragma unroll
    for (int hh = 0; hh < 16; ++hh) {
      b0[hh] = Ws[nt * 16 + lcol][hh + 16 * rHalf];
      b1[hh] = Ws[nt * 16 + lcol][32 + hh + 16 * rHalf];
    }
    const float bval = bia[nt * 16 + lcol];      // bias constant along M
    v8f acc;
    #pragma unroll
    for (int r = 0; r < 8; ++r) acc[r] = bval;
    acc = wmma_bf16(a0, b0, acc);
    acc = wmma_bf16(a1, b1, acc);

    #pragma unroll
    for (int r = 0; r < 8; ++r) {
      int gr = rowBase + wave * 16 + r + 8 * rHalf;   // flat n*L + l
      int n  = gr >> 11, l = gr & (LQ - 1);
      // layout (N, H, L, DH)
      Out[(((size_t)n * NH + h) * LQ + l) * DH + nt * 16 + lcol] = (bf16_t)acc[r];
    }
  }
}

// ---------------------------------------------------------------------------
// Kernel 2: flash attention per (n,h).  grid = (L/128, N*H); block = 256.
// Each wave owns a 16-row Q slab; K/V streamed through LDS in 64-row tiles
// via CDNA5 async global->LDS copies.
// ---------------------------------------------------------------------------
__global__ __launch_bounds__(256)
void attn_kernel(const bf16_t* __restrict__ Qb, const bf16_t* __restrict__ Kb,
                 const bf16_t* __restrict__ Vb, bf16_t* __restrict__ Ob)
{
  const int nh    = blockIdx.y;
  const int n     = nh / NH;
  const int h     = nh % NH;
  const int qBase = blockIdx.x * 128;

  const bf16_t* Qh = Qb + (size_t)nh * LQ * DH;
  const bf16_t* Kh = Kb + (size_t)nh * LQ * DH;
  const bf16_t* Vh = Vb + (size_t)nh * LQ * DH;

  __shared__ __align__(16) bf16_t Qs[128][72];     // 18KB
  __shared__ __align__(16) bf16_t Ks[64][72];      //  9KB
  __shared__ __align__(16) bf16_t Vs[64][72];      //  9KB
  __shared__ __align__(16) bf16_t Ps[8][16][72];   // 18KB, per-wave P scratch

  {
    const uint4* src = (const uint4*)(Qh + (size_t)qBase * DH);
    for (int i = threadIdx.x; i < 128 * 8; i += 256) {
      int r = i >> 3, c = i & 7;
      async_copy_b128(&Qs[r][c * 8], src + i);
    }
    wait_async0();
  }
  __syncthreads();

  const int wave  = threadIdx.x >> 5;
  const int lane  = threadIdx.x & 31;
  const int rHalf = lane >> 4;
  const int lcol  = lane & 15;

  v16bf qa0, qa1;
  #pragma unroll
  for (int hh = 0; hh < 16; ++hh) {
    int k = hh + 16 * rHalf;
    qa0[hh] = Qs[wave * 16 + lcol][k];
    qa1[hh] = Qs[wave * 16 + lcol][32 + k];
  }

  float mrow[8], lrow[8];
  v8f accO[4];
  #pragma unroll
  for (int r = 0; r < 8; ++r) { mrow[r] = -1e30f; lrow[r] = 0.f; }
  #pragma unroll
  for (int nt = 0; nt < 4; ++nt)
    #pragma unroll
    for (int r = 0; r < 8; ++r) accO[nt][r] = 0.f;

  const float SCALE = 0.03125f;   // 1/sqrt(E) = 1/32 (reference scales by sqrt(E))

  for (int kb = 0; kb < LQ; kb += 64) {
    __syncthreads();              // previous tile fully consumed
    {
      const uint4* ks = (const uint4*)(Kh + (size_t)kb * DH);
      const uint4* vs = (const uint4*)(Vh + (size_t)kb * DH);
      for (int i = threadIdx.x; i < 64 * 8; i += 256) {
        int r = i >> 3, c = i & 7;
        async_copy_b128(&Ks[r][c * 8], ks + i);
        async_copy_b128(&Vs[r][c * 8], vs + i);
      }
      wait_async0();
    }
    if (kb + 64 < LQ) {           // gfx1250: global_prefetch_b8 next tiles
      __builtin_prefetch(Kh + (size_t)(kb + 64) * DH + threadIdx.x * 16, 0, 1);
      __builtin_prefetch(Vh + (size_t)(kb + 64) * DH + threadIdx.x * 16, 0, 1);
    }
    __syncthreads();

    // S(16x64) = Q(16x64) @ K^T; B[kk][nn] = K[kb + ct*16 + nn][kk]
    v8f s[4];
    #pragma unroll
    for (int ct = 0; ct < 4; ++ct) {
      v16bf b0, b1;
      #pragma unroll
      for (int hh = 0; hh < 16; ++hh) {
        b0[hh] = Ks[ct * 16 + lcol][hh + 16 * rHalf];
        b1[hh] = Ks[ct * 16 + lcol][32 + hh + 16 * rHalf];
      }
      v8f sc;
      #pragma unroll
      for (int r = 0; r < 8; ++r) sc[r] = 0.f;
      sc = wmma_bf16(qa0, b0, sc);
      sc = wmma_bf16(qa1, b1, sc);
      s[ct] = sc;
    }
    #pragma unroll
    for (int ct = 0; ct < 4; ++ct)
      #pragma unroll
      for (int r = 0; r < 8; ++r) s[ct][r] *= SCALE;

    // online softmax; a row's 16 columns live across one 16-lane group
    float mnew[8], alpha[8], rsum[8];
    #pragma unroll
    for (int r = 0; r < 8; ++r) {
      float v = fmaxf(fmaxf(s[0][r], s[1][r]), fmaxf(s[2][r], s[3][r]));
      #pragma unroll
      for (int off = 1; off < 16; off <<= 1)
        v = fmaxf(v, __shfl_xor(v, off, 32));
      mnew[r]  = fmaxf(mrow[r], v);
      alpha[r] = __expf(mrow[r] - mnew[r]);
      mrow[r]  = mnew[r];
      rsum[r]  = 0.f;
    }
    #pragma unroll
    for (int ct = 0; ct < 4; ++ct) {
      #pragma unroll
      for (int r = 0; r < 8; ++r) {
        float p = __expf(s[ct][r] - mnew[r]);
        rsum[r] += p;
        // C-layout element (row r+8*rHalf, col ct*16+lcol) -> row-major LDS
        Ps[wave][r + 8 * rHalf][ct * 16 + lcol] = (bf16_t)p;
      }
    }
    #pragma unroll
    for (int r = 0; r < 8; ++r) {
      float v = rsum[r];
      #pragma unroll
      for (int off = 1; off < 16; off <<= 1)
        v += __shfl_xor(v, off, 32);
      lrow[r] = lrow[r] * alpha[r] + v;
    }
    #pragma unroll
    for (int nt = 0; nt < 4; ++nt)
      #pragma unroll
      for (int r = 0; r < 8; ++r) accO[nt][r] *= alpha[r];

    // re-read P as A-fragments (same-wave LDS ops are in-order; no barrier)
    v16bf pa0, pa1;
    #pragma unroll
    for (int hh = 0; hh < 16; ++hh) {
      int k = hh + 16 * rHalf;
      pa0[hh] = Ps[wave][lcol][k];
      pa1[hh] = Ps[wave][lcol][32 + k];
    }

    // accO += P(16x64) @ V(64x64); B[kk][nn] = V[kb+kk][nt*16+nn]
    #pragma unroll
    for (int nt = 0; nt < 4; ++nt) {
      v16bf b0, b1;
      #pragma unroll
      for (int hh = 0; hh < 16; ++hh) {
        b0[hh] = Vs[hh + 16 * rHalf][nt * 16 + lcol];
        b1[hh] = Vs[32 + hh + 16 * rHalf][nt * 16 + lcol];
      }
      accO[nt] = wmma_bf16(pa0, b0, accO[nt]);
      accO[nt] = wmma_bf16(pa1, b1, accO[nt]);
    }
  }

  // normalize, store attention output as bf16 in (N, L, E) layout
  #pragma unroll
  for (int nt = 0; nt < 4; ++nt) {
    #pragma unroll
    for (int r = 0; r < 8; ++r) {
      int ql = qBase + wave * 16 + r + 8 * rHalf;
      float o = accO[nt][r] / lrow[r];
      Ob[((size_t)n * LQ + ql) * EMB + h * DH + nt * 16 + lcol] = (bf16_t)o;
    }
  }
}

// ---------------------------------------------------------------------------
// Kernel 3: output projection.  out = Ob(4096x1024) @ Wo^T + bo, fp32 out.
// grid = (E/64, N*L/128); block tile 128(M) x 64(N), K stepped by 32.
// ---------------------------------------------------------------------------
__global__ __launch_bounds__(256)
void out_proj_kernel(const bf16_t* __restrict__ Ob, const float* __restrict__ Wo,
                     const float* __restrict__ bo, float* __restrict__ out)
{
  const int nBase = blockIdx.x * 64;
  const int mBase = blockIdx.y * 128;

  __shared__ __align__(16) bf16_t As[128][40];
  __shared__ __align__(16) bf16_t Bs[64][40];

  const int wave  = threadIdx.x >> 5;
  const int lane  = threadIdx.x & 31;
  const int rHalf = lane >> 4;
  const int lcol  = lane & 15;

  v8f acc[4];
  #pragma unroll
  for (int nt = 0; nt < 4; ++nt)
    #pragma unroll
    for (int r = 0; r < 8; ++r) acc[nt][r] = 0.f;

  for (int kb = 0; kb < EMB; kb += 32) {
    __syncthreads();
    // A tile via async global->LDS (128 rows x 32 cols bf16 = 4 x b128/row)
    for (int i = threadIdx.x; i < 128 * 4; i += 256) {
      int r = i >> 2, c = i & 3;
      async_copy_b128(&As[r][c * 8],
                      Ob + (size_t)(mBase + r) * EMB + kb + c * 8);
    }
    // Bs[nn][kk] = Wo[nBase+nn][kb+kk]  (B = Wo^T tile, fp32 -> bf16)
    for (int i = threadIdx.x; i < 64 * 32; i += 256) {
      int nn = i >> 5, kk = i & 31;
      Bs[nn][kk] = (bf16_t)Wo[(size_t)(nBase + nn) * EMB + kb + kk];
    }
    wait_async0();
    __syncthreads();

    v16bf a;
    #pragma unroll
    for (int hh = 0; hh < 16; ++hh) {
      int k = hh + 16 * rHalf;
      a[hh] = As[wave * 16 + lcol][k];
    }
    #pragma unroll
    for (int nt = 0; nt < 4; ++nt) {
      v16bf b;
      #pragma unroll
      for (int hh = 0; hh < 16; ++hh)
        b[hh] = Bs[nt * 16 + lcol][hh + 16 * rHalf];
      acc[nt] = wmma_bf16(a, b, acc[nt]);
    }
  }

  #pragma unroll
  for (int nt = 0; nt < 4; ++nt) {
    int col = nBase + nt * 16 + lcol;
    float bval = bo[col];
    #pragma unroll
    for (int r = 0; r < 8; ++r) {
      int row = mBase + wave * 16 + r + 8 * rHalf;
      out[(size_t)row * EMB + col] = acc[nt][r] + bval;
    }
  }
}

// ---------------------------------------------------------------------------
extern "C" void kernel_launch(void* const* d_in, const int* in_sizes, int n_in,
                              void* d_out, int out_size, void* d_ws, size_t ws_size,
                              hipStream_t stream) {
  const float* EPq = (const float*)d_in[0];
  const float* EPk = (const float*)d_in[1];
  const float* EPv = (const float*)d_in[2];
  const float* Wq  = (const float*)d_in[3];
  const float* bq  = (const float*)d_in[4];
  const float* Wk  = (const float*)d_in[5];
  const float* bk  = (const float*)d_in[6];
  const float* Wv  = (const float*)d_in[7];
  const float* bv  = (const float*)d_in[8];
  const float* Wo  = (const float*)d_in[9];
  const float* bo  = (const float*)d_in[10];
  float* out = (float*)d_out;

  // workspace: Qb | Kb | Vb (N,H,L,DH bf16) | Ob (N,L,E bf16) = 32 MB total
  const size_t headElems = (size_t)NB * NH * LQ * DH;
  bf16_t* Qb = (bf16_t*)d_ws;
  bf16_t* Kb = Qb + headElems;
  bf16_t* Vb = Kb + headElems;
  bf16_t* Ob = Vb + headElems;

  qkv_proj_kernel<<<dim3(NB * LQ / 128, NH, 3), 256, 0, stream>>>(
      EPq, EPk, EPv, Wq, bq, Wk, bk, Wv, bv, Qb, Kb, Vb);
  attn_kernel<<<dim3(LQ / 128, NB * NH), 256, 0, stream>>>(Qb, Kb, Vb, Ob);
  out_proj_kernel<<<dim3(EMB / 64, NB * LQ / 128), 256, 0, stream>>>(Ob, Wo, bo, out);
}
